// Encoder_73143293051532
// MI455X (gfx1250) — matrix operand order repaired and checked by reading, not compiled, not run on hardware
//
#include <hip/hip_runtime.h>

#define BB 32
#define TT 2048
#define UU 256
#define H4V 1024
#define HCV 512

// ---------------- device-only helpers ----------------
#if defined(__HIP_DEVICE_COMPILE__)

typedef __attribute__((ext_vector_type(16))) __bf16          bf16x16;
typedef __attribute__((ext_vector_type(8)))  float           f32x8;
typedef __attribute__((ext_vector_type(8)))  unsigned short  u16x8;
typedef __attribute__((ext_vector_type(16))) unsigned short  u16x16;
typedef __attribute__((ext_vector_type(4)))  unsigned int    u32x4;
typedef __attribute__((ext_vector_type(8)))  unsigned int    u32x8;

union BFrag { u16x16 u; bf16x16 b; };

__device__ __forceinline__ unsigned short f2bf(float f) {
    unsigned u = __float_as_uint(f);
    unsigned r = (u + 0x7FFFu + ((u >> 16) & 1u)) >> 16;   // RNE
    return (unsigned short)r;
}
__device__ __forceinline__ float bf2f(unsigned short s) {
    return __uint_as_float(((unsigned)s) << 16);
}
__device__ __forceinline__ float sigm(float v) {
    return 1.0f / (1.0f + __expf(-v));
}
__device__ __forceinline__ f32x8 wmma_bf16(const BFrag& a, const BFrag& b, f32x8 c) {
    return __builtin_amdgcn_wmma_f32_16x16x32_bf16(false, a.b, false, b.b, (short)0, c, false, false);
}
// A fragment for 16x32 bf16: lane<16 holds row (lane&15), K={0..7,16..23};
// lane>=16 same row, K={8..15,24..31}.  rowptr points at K=kt*32 of the row.
__device__ __forceinline__ void load_afrag(BFrag& a, const unsigned short* rowptr, int hi) {
    u16x8 lo = *(const u16x8*)(rowptr + hi);
    u16x8 hh = *(const u16x8*)(rowptr + 16 + hi);
#pragma unroll
    for (int i = 0; i < 8; ++i) { a.u[i] = lo[i]; a.u[8 + i] = hh[i]; }
}

// Tensor Data Mover: 2-D tile load global -> LDS.
// rows x row_elems (4-byte elements), contiguous (stride == row_elems).
// Issue from ONE wave only (EXEC ignored, per-wave issue).
__device__ __forceinline__ void tdm_load_2d(unsigned lds_byte_off, const void* gsrc,
                                            unsigned row_elems, unsigned rows) {
    unsigned long long ga = (unsigned long long)gsrc;
    u32x4 g0;
    g0[0] = 1u;                                          // count=1, user descriptor
    g0[1] = lds_byte_off;                                // lds_addr
    g0[2] = (unsigned)(ga & 0xFFFFFFFFu);                // global_addr[31:0]
    g0[3] = (unsigned)((ga >> 32) & 0x01FFFFFFu)         // global_addr[56:32]
            | 0x80000000u;                               // type=2 ("image")
    u32x8 g1;
    g1[0] = 0x00020000u;                                 // wg_mask=0, data_size=2 (4B)
    g1[1] = (row_elems & 0xFFFFu) << 16;                 // tensor_dim0[15:0]
    g1[2] = (row_elems >> 16) | ((rows & 0xFFFFu) << 16);// dim0[31:16] | dim1[15:0]
    g1[3] = (rows >> 16) | (row_elems << 16);            // dim1[31:16] | tile_dim0
    g1[4] = rows & 0xFFFFu;                              // tile_dim1 | tile_dim2=0
    g1[5] = row_elems;                                   // tensor_dim0_stride[31:0]
    g1[6] = 0u;                                          // stride0[47:32] | stride1[15:0]
    g1[7] = 0u;                                          // stride1[47:16]
    asm volatile("tensor_load_to_lds %0, %1" :: "s"(g0), "s"(g1) : "memory");
    __builtin_amdgcn_s_wait_tensorcnt(0);
}
#endif // __HIP_DEVICE_COMPILE__

// ---------------- weight packing ----------------
// B-fragment layout (dense bf16, 32x16 per k-tile): lane L holds column N=L&15,
// element e holds K=8*(L>>4)+(e&7)+16*(e>>3).  Frag storage: ((nt*KT + kt)*32 + lane)*16 + e.

__global__ void pack_w_frags(const float* __restrict__ W /*[512,1024]*/,
                             unsigned short* __restrict__ dst /*524288*/) {
#if defined(__HIP_DEVICE_COMPILE__)
    int idx = blockIdx.x * 256 + threadIdx.x;
    if (idx >= 64 * 16 * 32 * 16) return;
    int elem = idx & 15;
    int lane = (idx >> 4) & 31;
    int kt   = (idx >> 9) & 15;
    int nt   = idx >> 13;
    int k = kt * 32 + ((lane >> 4) << 3) + (elem & 7) + ((elem >> 3) << 4);
    int n = nt * 16 + (lane & 15);
    dst[idx] = f2bf(W[k * H4V + n]);
#endif
}

// U packed per (dir,half) with gate-column mapping:
// block-local ntile nt in [0,32): gate = nt>>3, global col = gate*256 + half*128 + (nt&7)*16 + e
__global__ void pack_u_frags(const float* __restrict__ Uw /*[256,1024]*/,
                             unsigned short* __restrict__ dst /*131072*/,
                             int half) {
#if defined(__HIP_DEVICE_COMPILE__)
    int idx = blockIdx.x * 256 + threadIdx.x;
    if (idx >= 32 * 8 * 32 * 16) return;
    int elem = idx & 15;
    int lane = (idx >> 4) & 31;
    int kt   = (idx >> 9) & 7;
    int nt   = idx >> 12;
    int gate = nt >> 3;
    int k = kt * 32 + ((lane >> 4) << 3) + (elem & 7) + ((elem >> 3) << 4);
    int n = gate * 256 + half * 128 + (nt & 7) * 16 + (lane & 15);
    dst[idx] = f2bf(Uw[k * H4V + n]);
#endif
}

// ---------------- xz = Abf(bf16 [65536,512]) * Wpack + bias -> xz fp32 [T,B,1024] ----------------
__global__ void xz_gemm(const unsigned short* __restrict__ Abf,
                        const unsigned short* __restrict__ Wp,
                        const float* __restrict__ bias,
                        float* __restrict__ xz) {
#if defined(__HIP_DEVICE_COMPILE__)
    int tid  = blockIdx.x * blockDim.x + threadIdx.x;
    int wid  = tid >> 5;
    int lane = tid & 31;
    int mtile = wid >> 4;      // [0,4096)
    int ng    = wid & 15;      // group of 4 ntiles
    int hi = (lane & 16) ? 8 : 0;
    const unsigned short* arow = Abf + (size_t)(mtile * 16 + (lane & 15)) * HCV;

    f32x8 acc[4] = {};
    for (int kt = 0; kt < 16; ++kt) {
        if (kt < 14) __builtin_prefetch(arow + (kt + 2) * 32, 0, 1);
        BFrag a;
        load_afrag(a, arow + kt * 32, hi);
#pragma unroll
        for (int q = 0; q < 4; ++q) {
            int nt = ng * 4 + q;
            BFrag b;
            b.u = *(const u16x16*)(Wp + ((size_t)(nt * 16 + kt) * 32 + lane) * 16);
            acc[q] = wmma_bf16(a, b, acc[q]);
        }
    }
    int rb = hi; // row offset 8 for lanes>=16
#pragma unroll
    for (int q = 0; q < 4; ++q) {
        int n = (ng * 4 + q) * 16 + (lane & 15);
        float bn = bias[n];
#pragma unroll
        for (int r = 0; r < 8; ++r) {
            int m = mtile * 16 + rb + r;
            int b_ = m >> 11;          // batch
            int t  = m & (TT - 1);     // time
            xz[((size_t)t * BB + b_) * H4V + n] = acc[q][r] + bn;
        }
    }
#endif
}

// ---------------- recurrent LSTM scan ----------------
// grid = 4 blocks: blockIdx.x = dir*2 + half.  256 threads (8 waves).
// LDS: Upack slice (256KB) + h buffer (32x264 bf16) + z staging (32x520 bf16) + x column.
#define SM_U   0
#define SM_H   262144
#define SM_Z   (262144 + 16896)
#define SM_X   (262144 + 16896 + 33280)
#define SM_TOT (262144 + 16896 + 33280 + 128)

__global__ void lstm_scan(const float* __restrict__ xzf, const float* __restrict__ xzb,
                          const float* __restrict__ x0,
                          const float* __restrict__ W0fp, const float* __restrict__ b0fp,
                          const float* __restrict__ W0bp, const float* __restrict__ b0bp,
                          const unsigned short* __restrict__ Upk,
                          float* __restrict__ states,
                          float* __restrict__ outf, unsigned short* __restrict__ outbf,
                          unsigned short* __restrict__ xch, int* __restrict__ flags) {
#if defined(__HIP_DEVICE_COMPILE__)
    extern __shared__ char smem[];
    unsigned short* sU  = (unsigned short*)(smem + SM_U);
    unsigned short* shb = (unsigned short*)(smem + SM_H);   // h, row stride 264
    unsigned short* szb = (unsigned short*)(smem + SM_Z);   // z, row stride 520
    float*          sx  = (float*)(smem + SM_X);

    const int tid  = threadIdx.x;
    const int dir  = blockIdx.x >> 1;
    const int half = blockIdx.x & 1;

    const float* xz = dir ? xzb : xzf;
    const float* W0 = dir ? W0bp : W0fp;
    const float* b0 = dir ? b0bp : b0fp;

    // resident U slice -> LDS (256 KB) via Tensor Data Mover (wave 0 issues)
    const unsigned short* Up = Upk + (size_t)(dir * 2 + half) * 131072;
    if (tid < 32)
        tdm_load_2d(SM_U, Up, /*row_elems(4B)=*/16384, /*rows=*/4);
    (void)sU;

    float* hst = states + dir * 16384;       // [32,256]
    float* cst = hst + 8192;

    // full h0 -> LDS (bf16)
    for (int i = tid; i < 8192; i += 256)
        shb[(i >> 8) * 264 + (i & 255)] = f2bf(hst[i]);

    // c state: thread owns (r0 + 2k, j) for k=0..15
    const int j  = tid & 127;
    const int r0 = tid >> 7;
    float creg[16], hreg[16];
#pragma unroll
    for (int k = 0; k < 16; ++k) {
        creg[k] = cst[(r0 + 2 * k) * 256 + half * 128 + j];
        hreg[k] = 0.0f;
    }

    const int lane = tid & 31;
    const int w    = tid >> 5;           // wave 0..7, ntiles 4w..4w+3
    const int hi   = (lane & 16) ? 8 : 0;
    const int rb   = hi;

    float w0r[4], b0r[4];
    int   gcol[4];
#pragma unroll
    for (int q = 0; q < 4; ++q) {
        int nt = 4 * w + q;
        int g  = (nt >> 3) * 256 + half * 128 + (nt & 7) * 16 + (lane & 15);
        gcol[q] = g;
        w0r[q] = x0 ? W0[g] : 0.0f;
        b0r[q] = x0 ? b0[g] : 0.0f;
    }

    const int self = dir * 2 + half, part = dir * 2 + (half ^ 1);
    unsigned short*       xs = xch + self * 4096;
    const unsigned short* xp = xch + part * 4096;
    const unsigned long long xpbase = (unsigned long long)xp;

    for (int step = 0; step < TT; ++step) {
        int t = dir ? (TT - 1 - step) : step;
        if (x0 && tid < 32) sx[tid] = x0[tid * TT + t];
        __syncthreads();

        // z = h @ U  (2 mtiles x 4 ntiles x 8 ktiles of wmma per wave)
        f32x8 acc[2][4] = {};
        for (int kt = 0; kt < 8; ++kt) {
            BFrag a0, a1;
            load_afrag(a0, shb + (lane & 15) * 264 + kt * 32, hi);
            load_afrag(a1, shb + (16 + (lane & 15)) * 264 + kt * 32, hi);
#pragma unroll
            for (int q = 0; q < 4; ++q) {
                int nt = 4 * w + q;
                BFrag bfr;
                bfr.u = *(const u16x16*)(sU + ((size_t)(nt * 8 + kt) * 32 + lane) * 16);
                acc[0][q] = wmma_bf16(a0, bfr, acc[0][q]);
                acc[1][q] = wmma_bf16(a1, bfr, acc[1][q]);
            }
        }

        // z += xz (or rank-1 layer-0 projection), stage to LDS
#pragma unroll
        for (int q = 0; q < 4; ++q) {
            int nl = (4 * w + q) * 16 + (lane & 15);
            int g  = gcol[q];
#pragma unroll
            for (int mt = 0; mt < 2; ++mt) {
#pragma unroll
                for (int rr = 0; rr < 8; ++rr) {
                    int row = mt * 16 + rb + rr;
                    float z = acc[mt][q][rr];
                    z += x0 ? (sx[row] * w0r[q] + b0r[q])
                            : xz[((size_t)t * BB + row) * H4V + g];
                    szb[row * 520 + nl] = f2bf(z);
                }
            }
        }
        __syncthreads();

        // gate combine: i,f,g,o at block-local cols j, 128+j, 256+j, 384+j
#pragma unroll
        for (int k = 0; k < 16; ++k) {
            int r = r0 + 2 * k;
            float zi = bf2f(szb[r * 520 + j]);
            float zf = bf2f(szb[r * 520 + 128 + j]);
            float zg = bf2f(szb[r * 520 + 256 + j]);
            float zo = bf2f(szb[r * 520 + 384 + j]);
            float c = sigm(zf) * creg[k] + sigm(zi) * tanhf(zg);
            creg[k] = c;
            float h = sigm(zo) * tanhf(c);
            hreg[k] = h;
            unsigned short hb = f2bf(h);
            shb[r * 264 + half * 128 + j] = hb;
            xs[r * 128 + j] = hb;
            size_t oi = ((size_t)r * TT + t) * HCV + dir * UU + half * 128 + j;
            if (outf)  outf[oi]  = h;
            if (outbf) outbf[oi] = hb;
        }

        // publish own half, wait for partner half
        __threadfence();
        __syncthreads();
        if (tid == 0)
            __hip_atomic_store(flags + self, step + 1, __ATOMIC_RELEASE, __HIP_MEMORY_SCOPE_AGENT);
        while (__hip_atomic_load(flags + part, __ATOMIC_ACQUIRE, __HIP_MEMORY_SCOPE_AGENT) < step + 1)
            __builtin_amdgcn_s_sleep(2);

        // async global -> LDS copy of partner half (8KB, 16B per lane per issue)
#pragma unroll
        for (int c = tid; c < 512; c += 256) {
            int r  = c >> 4;
            int ck = c & 15;
            unsigned voff = (unsigned)(r * 256 + ck * 16);
            unsigned ldst = (unsigned)(SM_H + r * 528 + (half ^ 1) * 256 + ck * 16);
            asm volatile("global_load_async_to_lds_b128 %0, %1, %2"
                         :: "v"(ldst), "v"(voff), "s"(xpbase) : "memory");
        }
        asm volatile("s_wait_asynccnt 0x0" ::: "memory");
        // loop-top __syncthreads() orders these LDS writes vs next step's reads
    }

    // final states for this (dir, half) slice
#pragma unroll
    for (int k = 0; k < 16; ++k) {
        int r = r0 + 2 * k;
        hst[r * 256 + half * 128 + j] = hreg[k];
        cst[r * 256 + half * 128 + j] = creg[k];
    }
#endif
}

// ---------------- host launcher ----------------
extern "C" void kernel_launch(void* const* d_in, const int* in_sizes, int n_in,
                              void* d_out, int out_size, void* d_ws, size_t ws_size,
                              hipStream_t stream) {
    (void)in_sizes; (void)n_in; (void)out_size; (void)ws_size;

    const float* x   = (const float*)d_in[0];
    const float* W0f = (const float*)d_in[1];
    const float* U0f = (const float*)d_in[2];
    const float* b0f = (const float*)d_in[3];
    const float* W0b = (const float*)d_in[4];
    const float* U0b = (const float*)d_in[5];
    const float* b0b = (const float*)d_in[6];
    const float* Wf  = (const float*)d_in[7];
    const float* Uf  = (const float*)d_in[8];
    const float* bf_ = (const float*)d_in[9];
    const float* Wb  = (const float*)d_in[10];
    const float* Ub  = (const float*)d_in[11];
    const float* bb_ = (const float*)d_in[12];

    char* ws = (char*)d_ws;
    size_t off = 0;
    auto alloc = [&](size_t bytes) { size_t o = off; off += (bytes + 255) & ~(size_t)255; return o; };

    float*          xzf = (float*)(ws + alloc((size_t)TT * BB * H4V * 4));
    float*          xzb = (float*)(ws + alloc((size_t)TT * BB * H4V * 4));
    unsigned short* bfA = (unsigned short*)(ws + alloc((size_t)BB * TT * HCV * 2));
    unsigned short* bfB = (unsigned short*)(ws + alloc((size_t)BB * TT * HCV * 2));
    unsigned short* UPK = (unsigned short*)(ws + alloc((size_t)12 * 131072 * 2));
    unsigned short* WPK = (unsigned short*)(ws + alloc((size_t)4 * 524288 * 2));
    float*          STS = (float*)(ws + alloc((size_t)32768 * 4));
    unsigned short* XCH = (unsigned short*)(ws + alloc((size_t)4 * 4096 * 2));
    int*            FLG = (int*)(ws + alloc(64));

    hipMemsetAsync(STS, 0, 32768 * 4, stream);
    hipMemsetAsync(FLG, 0, 12 * sizeof(int), stream);

    // pack recurrent weights: slice index = layer*4 + dir*2 + half
    for (int layer = 0; layer < 3; ++layer)
        for (int dir = 0; dir < 2; ++dir)
            for (int half = 0; half < 2; ++half) {
                const float* us = (layer == 0)
                                      ? (dir ? U0b : U0f)
                                      : ((dir ? Ub : Uf) + (size_t)(layer - 1) * UU * H4V);
                pack_u_frags<<<512, 256, 0, stream>>>(
                    us, UPK + (size_t)(layer * 4 + dir * 2 + half) * 131072, half);
            }
    // pack input-projection weights for layers 1..2: slice = (l-1)*2 + dir
    for (int l = 1; l < 3; ++l)
        for (int dir = 0; dir < 2; ++dir) {
            const float* wsrc = (dir ? Wb : Wf) + (size_t)(l - 1) * HCV * H4V;
            pack_w_frags<<<2048, 256, 0, stream>>>(
                wsrc, WPK + (size_t)((l - 1) * 2 + dir) * 524288);
        }

    // layer 0: rank-1 projection folded into scan
    lstm_scan<<<4, 256, SM_TOT, stream>>>(
        nullptr, nullptr, x, W0f, b0f, W0b, b0b,
        UPK, STS, nullptr, bfA, XCH, FLG);

    // layer 1
    xz_gemm<<<8192, 256, 0, stream>>>(bfA, WPK + (size_t)0 * 524288, bf_ + 0,    xzf);
    xz_gemm<<<8192, 256, 0, stream>>>(bfA, WPK + (size_t)1 * 524288, bb_ + 0,    xzb);
    lstm_scan<<<4, 256, SM_TOT, stream>>>(
        xzf, xzb, nullptr, nullptr, nullptr, nullptr, nullptr,
        UPK + (size_t)4 * 131072, STS, nullptr, bfB, XCH, FLG + 4);

    // layer 2 (final) -> d_out
    xz_gemm<<<8192, 256, 0, stream>>>(bfB, WPK + (size_t)2 * 524288, bf_ + H4V,  xzf);
    xz_gemm<<<8192, 256, 0, stream>>>(bfB, WPK + (size_t)3 * 524288, bb_ + H4V,  xzb);
    lstm_scan<<<4, 256, SM_TOT, stream>>>(
        xzf, xzb, nullptr, nullptr, nullptr, nullptr, nullptr,
        UPK + (size_t)8 * 131072, STS, (float*)d_out, nullptr, XCH, FLG + 8);

    // states (hf, cf, hb, cb) appended after out
    hipMemcpyAsync((float*)d_out + (size_t)BB * TT * HCV, STS, 32768 * 4,
                   hipMemcpyDeviceToDevice, stream);
}
